// Sequence_18219251269654
// MI455X (gfx1250) — matrix operand order repaired and checked by reading, not compiled, not run on hardware
//
#include <hip/hip_runtime.h>
#include <cmath>

#define HID    51
#define KPAD   64     // K = HID (h) + x col + bias col, padded to 64
#define XCOL   51     // A column carrying x_t
#define ONECOL 52     // A column pinned to 1.0 (bias row of B)
#define BROWS  16     // batch rows per block (one 16-row WMMA M strip)
#define LMAX   1000

typedef __attribute__((ext_vector_type(16))) _Float16 v16h;
typedef __attribute__((ext_vector_type(8)))  _Float16 v8h;
typedef __attribute__((ext_vector_type(8)))  float    v8f;

__device__ __forceinline__ float fast_tanh(float v) {
#if __has_builtin(__builtin_amdgcn_tanhf)
    return __builtin_amdgcn_tanhf(v);   // V_TANH_F32 (gfx1250 transcendental)
#else
    return tanhf(v);
#endif
}

// Split workgroup barrier (CDNA5): signal after LDS writes drain, wait at the
// top of the next trip.  Lets global output stores + loop bookkeeping overlap
// with other waves' barrier arrival.
#define BAR_SIGNAL() asm volatile("s_wait_dscnt 0x0\n\ts_barrier_signal -1" ::: "memory")
#define BAR_WAIT()   asm volatile("s_barrier_wait -1" ::: "memory")

__launch_bounds__(128, 1)
__global__ void lstm_scan_wmma(const float* __restrict__ x,
                               const float* __restrict__ W_w,
                               const float* __restrict__ W_b,
                               const float* __restrict__ U_w,
                               const float* __restrict__ U_b,
                               const float* __restrict__ lin_w,
                               const float* __restrict__ lin_b,
                               float* __restrict__ out,
                               int L)
{
    __shared__ __align__(16) _Float16 x_lds[BROWS][LMAX];     // 32 KB, once
    // Double-buffered recurrent state: trip ts reads [ts&1], writes [ts&1^1].
    // One barrier per trip is then sufficient (no read/write overlap).
    __shared__ __align__(16) _Float16 h_lds[2][BROWS][KPAD];  // [h | x | 1 | 0]

    const int tid  = threadIdx.x;
    const int wave = tid >> 5;          // u-slice: columns wave*16 .. wave*16+15
    const int lane = tid & 31;
    const int m16  = lane & 15;
    const int hi   = lane >> 4;
    const int b0   = blockIdx.x * BROWS;
    const int u    = wave * 16 + m16;   // column index inside each gate (0..63)

    const float linb = lin_b[0];

    // ---- one-time: pack B fragments (fp16) --------------------------------
    // B rows: K<51 -> U_w.T ; K==51 -> W_w (x term) ; K==52 -> W_b+U_b (bias,
    // paired with A column 52 == 1.0) ; rest 0.  Padded column u==51 of
    // gate 0 carries [lin_w | 0 | lin_b]: the output projection rides the
    // same WMMAs for free.  Bias-in-K lets every gate start from an
    // inline-0 C operand (no accumulator-init copies).
    v16h bfrag[4][2];
#pragma unroll
    for (int g = 0; g < 4; ++g) {
        int  j     = g * HID + u;
        bool valid = (u < HID);
        bool isout = (g == 0) && (u == XCOL);
#pragma unroll
        for (int kc = 0; kc < 2; ++kc) {
            v16h b;
#pragma unroll
            for (int e = 0; e < 16; ++e) {
                int   K = kc * 32 + ((e < 8) ? (hi * 8 + e) : (16 + hi * 8 + (e - 8)));
                float v = 0.0f;
                if (valid) {
                    if (K < HID)          v = U_w[j * HID + K];
                    else if (K == XCOL)   v = W_w[j];
                    else if (K == ONECOL) v = W_b[j] + U_b[j];
                } else if (isout) {
                    if (K < HID)          v = lin_w[K];
                    else if (K == ONECOL) v = linb;
                }
                b[e] = (_Float16)v;
            }
            bfrag[g][kc] = b;
        }
    }

    // ---- one-time: stage x strip (fp16) + init both state buffers ---------
    for (int r = 0; r < BROWS; ++r)
        for (int t = tid; t < L; t += 128)
            x_lds[r][t] = (_Float16)x[(size_t)(b0 + r) * L + t];
    for (int idx = tid; idx < 2 * BROWS * KPAD; idx += 128) {
        int col = idx & 63;
        h_lds[0][0][idx] = (col == ONECOL) ? (_Float16)1.0f : (_Float16)0.0f;
    }
    __syncthreads();                                   // staging visible
    if (tid < BROWS) h_lds[0][tid][XCOL] = x_lds[tid][0];  // buf0 col51 <- x_0
    BAR_SIGNAL();                                      // release trip 0

    v8f creg;                            // cell state for this wave's slice
#pragma unroll
    for (int e = 0; e < 8; ++e) creg[e] = 0.0f;

    // ======================= recurrent scan ================================
    // Trip ts consumes h_{ts-1}: yields out[:, ts-1] (projection column) and
    // produces h_ts/c_ts.  L+1 trips; trip L only harvests the last output.
    for (int ts = 0; ts <= L; ++ts) {
        const int rb = ts & 1;
        BAR_WAIT();                      // read buffer ready

        // A-fragments from LDS (16-bit A 16x32 layout), shared by all gates
        const _Float16* hrow = &h_lds[rb][m16][0];
        v16h a0, a1;
        {
            v8h p0 = *(const v8h*)&hrow[hi * 8];
            v8h p1 = *(const v8h*)&hrow[16 + hi * 8];
            v8h p2 = *(const v8h*)&hrow[32 + hi * 8];
            v8h p3 = *(const v8h*)&hrow[48 + hi * 8];
#pragma unroll
            for (int e = 0; e < 8; ++e) {
                a0[e] = p0[e]; a0[8 + e] = p1[e];
                a1[e] = p2[e]; a1[8 + e] = p3[e];
            }
        }

        // gates = [h, x_t, 1] @ B : 4 gates x 2 K-chunks, C starts inline-0
        v8f acc[4];
#pragma unroll
        for (int g = 0; g < 4; ++g) {
            v8f zero = {};
            acc[g] = __builtin_amdgcn_wmma_f32_16x16x32_f16(
                         false, a0, false, bfrag[g][0], (short)0, zero, false, false);
            acc[g] = __builtin_amdgcn_wmma_f32_16x16x32_f16(
                         false, a1, false, bfrag[g][1], (short)0, acc[g], false, false);
        }

        if (ts < L) {
            // Branch-free cell update for all lanes (pad lanes -> garbage,
            // never stored); single exec-masked region for the h stores.
            float hv[8];
#pragma unroll
            for (int r = 0; r < 8; ++r) {
                float c = acc[1][r] * creg[r] + acc[0][r] * acc[2][r];
                creg[r] = c;
                hv[r]   = acc[3][r] * fast_tanh(c);
            }
            if (u < HID) {
#pragma unroll
                for (int r = 0; r < 8; ++r)
                    h_lds[rb ^ 1][r + hi * 8][u] = (_Float16)hv[r];
            }
            if (tid < BROWS)
                h_lds[rb ^ 1][tid][XCOL] = (ts + 1 < L) ? x_lds[tid][ts + 1]
                                                        : (_Float16)0.0f;
            BAR_SIGNAL();                // write buffer published
        }

        // Harvest out[:, ts-1] from gate-0's projection column (u==51).
        // After the signal: these global stores drain off the barrier path.
        if (ts > 0 && u == XCOL) {
            float* op = out + (size_t)(b0 + hi * 8) * L + (ts - 1);
#pragma unroll
            for (int r = 0; r < 8; ++r)
                op[(size_t)r * L] = acc[0][r];
        }
    }
}

extern "C" void kernel_launch(void* const* d_in, const int* in_sizes, int n_in,
                              void* d_out, int out_size, void* d_ws, size_t ws_size,
                              hipStream_t stream) {
    const float* x     = (const float*)d_in[0];
    const float* W_w   = (const float*)d_in[1];
    const float* W_b   = (const float*)d_in[2];
    const float* U_w   = (const float*)d_in[3];
    const float* U_b   = (const float*)d_in[4];
    const float* lin_w = (const float*)d_in[5];
    const float* lin_b = (const float*)d_in[6];
    const int B = 1024;
    int L = in_sizes[0] / B;
    if (L > LMAX) L = LMAX;
    lstm_scan_wmma<<<B / BROWS, 128, 0, stream>>>(
        x, W_w, W_b, U_w, U_b, lin_w, lin_b, (float*)d_out, L);
}
